// TFPerformerAttention_52166672778206
// MI455X (gfx1250) — compile-verified, matrix operand order checked by reading, not run on hardware
//
#include <hip/hip_runtime.h>
#include <hip/hip_bf16.h>
#include <stdint.h>

// ---------------------------------------------------------------------------
// Performer (FAVOR+) causal attention, fused chunked kernel for gfx1250.
//   B=4, L=2048, H=16, d_head=64, m=256, chunk C=64.
// One workgroup (8 wave32) per (b,h); 32 sequential chunks.
// All GEMMs via v_wmma_f32_16x16x32_bf16; operand fragments are contiguous
// ds_load_b128 pairs, and fragments are shared across 2-8 WMMAs per load
// (proj: B shared over 8; A/O-inter fused: A shared over 4; S-update: A
// shared over 4).  Global->LDS uses double-buffered
// global_load_async_to_lds_b128 (ASYNCcnt pipeline).  S stays in f32 VGPRs.
// ---------------------------------------------------------------------------

typedef __bf16 bf16;
typedef __attribute__((ext_vector_type(16))) __bf16 v16bf;
typedef __attribute__((ext_vector_type(8)))  __bf16 v8bf;
typedef __attribute__((ext_vector_type(8)))  float  v8f;

#define WMMA_BF16(a, b, c) \
  __builtin_amdgcn_wmma_f32_16x16x32_bf16(false, (a), false, (b), (short)0, (c), false, false)

constexpr int   Bn = 4, Ln = 2048, Hn = 16, Dh = 64, Mf = 256, Cc = 64, NCH = Ln / Cc, DM = 1024;
constexpr float KSCALE = 0.17677669529663687f;  // dm^-0.25
constexpr float SCALE2 = 0.03125f;              // dm^-0.5
constexpr float NC_F   = 0.0625f;               // m^-0.5
constexpr float KEPS   = 1e-4f;
constexpr float NSTAB  = 1e-6f;

// ---- dynamic LDS carve ----------------------------------------------------
constexpr int OFF_RFT = 0;                   // bf16 [256][64]  RF^T  (B: proj)
constexpr int OFF_QP  = OFF_RFT + 32768;     // bf16 [64][256]  q'    (A: A,O)
constexpr int OFF_KP  = OFF_QP  + 32768;     // bf16 [64][256]  k'    (B: A=Q'K'^T)
constexpr int OFF_KPT = OFF_KP  + 32768;     // bf16 [256][64]  k'^T  (A: S-update; Z sums)
constexpr int OFF_SBT = OFF_KPT + 32768;     // bf16 [64][256]  S^T   (B: O-inter)
constexpr int OFF_QB  = OFF_SBT + 32768;     // bf16 [64][64]   scaled q chunk (A: proj)
constexpr int OFF_KB  = OFF_QB  + 8192;      // bf16 [64][64]   scaled k chunk (A: proj)
constexpr int OFF_VBT = OFF_KB  + 8192;      // bf16 [64][64]   V^T   (B: O-intra, S-update)
constexpr int OFF_AB  = OFF_VBT + 8192;      // bf16 [64][64]   masked A (A: O-intra)
constexpr int OFF_A   = OFF_AB  + 8192;      // f32  [64][64]   raw A
constexpr int OFF_Z   = OFF_A   + 16384;     // f32  [256]
constexpr int OFF_HK  = OFF_Z   + 1024;      // f32  [64]
constexpr int OFF_DEN = OFF_HK  + 256;       // f32  [64]
constexpr int OFF_STG = OFF_DEN + 256;       // f32 staging: 2 bufs x (Q,K,V) x 16KB
constexpr int STG_MAT = 16384;
constexpr int STG_BUF = 3 * STG_MAT;         // 49152
constexpr int SMEM_BYTES = OFF_STG + 2 * STG_BUF;  // 312832 B (< 320 KB WGP LDS)

// ---- unified WMMA fragment loader -----------------------------------------
union FragU { struct { v8bf lo, hi; } h; v16bf v; };

__device__ __forceinline__ v16bf frag(const bf16* base, int ld, int a0, int k0, int lane) {
  const bf16* p = base + (a0 + (lane & 15)) * ld + k0 + ((lane >> 4) << 3);
  FragU f;
  f.h.lo = *(const v8bf*)(p);
  f.h.hi = *(const v8bf*)(p + 16);
  return f.v;
}

// ---- async global -> LDS (ASYNCcnt) ---------------------------------------
#define ASYNC_B128(ldsv, voff, sbase, OFFS)                                 \
  asm volatile("global_load_async_to_lds_b128 %0, %1, %2 offset:" OFFS      \
               :: "v"(ldsv), "v"(voff), "s"(sbase) : "memory")

__device__ __forceinline__ void issue_chunk_async(uint32_t ldsQ, uint32_t ldsK, uint32_t ldsV,
                                                  const float* gQ, const float* gK,
                                                  const float* gV, uint32_t voff) {
  ASYNC_B128(ldsQ, voff, gQ, "0");   ASYNC_B128(ldsQ, voff, gQ, "16");
  ASYNC_B128(ldsQ, voff, gQ, "32");  ASYNC_B128(ldsQ, voff, gQ, "48");
  ASYNC_B128(ldsK, voff, gK, "0");   ASYNC_B128(ldsK, voff, gK, "16");
  ASYNC_B128(ldsK, voff, gK, "32");  ASYNC_B128(ldsK, voff, gK, "48");
  ASYNC_B128(ldsV, voff, gV, "0");   ASYNC_B128(ldsV, voff, gV, "16");
  ASYNC_B128(ldsV, voff, gV, "32");  ASYNC_B128(ldsV, voff, gV, "48");
}

// ---- pass 1: global min ||k_row||^2  (-> k_stab) --------------------------
__global__ void performer_init_ws(unsigned* minbits) { *minbits = 0x7F7FFFFFu; }

__global__ __launch_bounds__(256) void performer_kstab(const float* __restrict__ Kg,
                                                       unsigned* __restrict__ minbits) {
  __shared__ float red[256];
  const int tid = threadIdx.x;
  const int nrows = Bn * Ln * Hn;
  float lmin = 3.0e38f;
  for (int r = blockIdx.x * blockDim.x + tid; r < nrows; r += gridDim.x * blockDim.x) {
    const float* p = Kg + (size_t)r * Dh;
    float s = 0.f;
#pragma unroll 8
    for (int i = 0; i < Dh; ++i) s += p[i] * p[i];
    lmin = fminf(lmin, s);
  }
  red[tid] = lmin;
  __syncthreads();
  for (int off = 128; off > 0; off >>= 1) {
    if (tid < off) red[tid] = fminf(red[tid], red[tid + off]);
    __syncthreads();
  }
  if (tid == 0) atomicMin(minbits, __float_as_uint(red[0]));
}

// ---- pass 2: fused projection + feature map + chunked causal attention ----
__global__ __launch_bounds__(256, 1)
void performer_chunk(const float* __restrict__ Q, const float* __restrict__ Kg,
                     const float* __restrict__ V, const float* __restrict__ RF,
                     const unsigned* __restrict__ minbits, float* __restrict__ Out) {
  extern __shared__ char smem[];
  bf16*  sRFT = (bf16*)(smem + OFF_RFT);
  bf16*  sQP  = (bf16*)(smem + OFF_QP);
  bf16*  sKP  = (bf16*)(smem + OFF_KP);
  bf16*  sKPT = (bf16*)(smem + OFF_KPT);
  bf16*  sSbT = (bf16*)(smem + OFF_SBT);
  bf16*  sQb  = (bf16*)(smem + OFF_QB);
  bf16*  sKb  = (bf16*)(smem + OFF_KB);
  bf16*  sVbT = (bf16*)(smem + OFF_VBT);
  bf16*  sAb  = (bf16*)(smem + OFF_AB);
  float* sA   = (float*)(smem + OFF_A);
  float* sZ   = (float*)(smem + OFF_Z);
  float* sHK  = (float*)(smem + OFF_HK);
  float* sDen = (float*)(smem + OFF_DEN);

  const int tid  = threadIdx.x;
  const int lane = tid & 31;
  const int wave = tid >> 5;
  const int b    = blockIdx.x >> 4;
  const int hh   = blockIdx.x & 15;

  const float minssq = __uint_as_float(*minbits);
  const float khoff  = 0.5f * SCALE2 * minssq;     // hk = -0.5*S2*ssq + khoff

  // random features -> bf16 LDS, transposed [m][d]
  for (int i = tid; i < Dh * Mf; i += 256) {
    const int dd = i & 63, mm = i >> 6;
    sRFT[i] = (bf16)RF[dd * Mf + mm];
  }
  sZ[tid] = 0.f;

  v8f Sacc[8] = {};                                // S[256,64]: j = mig*4+di

  const int row1 = tid & 63;
  const int q4   = tid >> 6;
  const int cbase = row1 * Dh + q4 * 16;
  const uint32_t voffb = (uint32_t)((row1 * DM + q4 * 16) * 4);
  const uint32_t ldsStg0 = (uint32_t)(uintptr_t)(smem + OFF_STG) + (uint32_t)(cbase * 4);

  // per-wave tile geometry
  const int riO  = wave >> 1;                      // row tile for A/O GEMMs
  const int ci0  = (wave & 1) * 2;                 // two column tiles
  const int ci1  = ci0 + 1;
  const int laneA = lane & 15;
  const int laneR = (lane >> 4) << 3;

  // prologue: async-load chunk 0 into staging buffer 0
  {
    const size_t gb = (size_t)(b * Ln) * DM + hh * Dh;
    issue_chunk_async(ldsStg0, ldsStg0 + STG_MAT, ldsStg0 + 2 * STG_MAT,
                      Q + gb, Kg + gb, V + gb, voffb);
  }
  __syncthreads();

  for (int c = 0; c < NCH; ++c) {
    const int l0  = c * Cc;
    const int buf = c & 1;

    // --- issue async loads for chunk c+1 (other buffer), then drain c ---
    if (c + 1 < NCH) {
      const uint32_t ldsN = ldsStg0 + (uint32_t)((buf ^ 1) * STG_BUF);
      const size_t gb = (size_t)(b * Ln + (c + 1) * Cc) * DM + hh * Dh;
      issue_chunk_async(ldsN, ldsN + STG_MAT, ldsN + 2 * STG_MAT,
                        Q + gb, Kg + gb, V + gb, voffb);
      asm volatile("s_wait_asynccnt 0xc" ::: "memory");
    } else {
      asm volatile("s_wait_asynccnt 0x0" ::: "memory");
    }

    // --- export S_prev (transposed bf16) for O GEMM; zero ssq scratch ---
    if (tid < 64) sHK[tid] = 0.f;
#pragma unroll
    for (int j = 0; j < 8; ++j) {
      const int mi = wave * 2 + (j >> 2), di = j & 3;
      const int col = di * 16 + laneA;
      const int r0  = mi * 16 + laneR;
      v8bf pk;
#pragma unroll
      for (int r = 0; r < 8; ++r) pk[r] = (bf16)Sacc[j][r];
      *(v8bf*)(sSbT + col * Mf + r0) = pk;
    }

    // --- convert staged chunk: scale q/k, ssq(k), V transposed ---
    {
      const float* stQ = (const float*)(smem + OFF_STG + buf * STG_BUF);
      const float* stK = stQ + STG_MAT / 4;
      const float* stV = stQ + 2 * STG_MAT / 4;
      float ssq = 0.f;
      v8bf q0, q1, k0, k1;
#pragma unroll
      for (int i = 0; i < 8; ++i) {
        const float kraw = stK[cbase + i];
        ssq += kraw * kraw;
        q0[i] = (bf16)(stQ[cbase + i] * KSCALE);
        k0[i] = (bf16)(kraw * KSCALE);
        sVbT[(q4 * 16 + i) * Dh + row1] = (bf16)stV[cbase + i];
      }
#pragma unroll
      for (int i = 0; i < 8; ++i) {
        const float kraw = stK[cbase + 8 + i];
        ssq += kraw * kraw;
        q1[i] = (bf16)(stQ[cbase + 8 + i] * KSCALE);
        k1[i] = (bf16)(kraw * KSCALE);
        sVbT[(q4 * 16 + 8 + i) * Dh + row1] = (bf16)stV[cbase + 8 + i];
      }
      *(v8bf*)(sQb + cbase) = q0;  *(v8bf*)(sQb + cbase + 8) = q1;
      *(v8bf*)(sKb + cbase) = k0;  *(v8bf*)(sKb + cbase + 8) = k1;
      atomicAdd(&sHK[row1], ssq);
    }
    __syncthreads();

    // --- projection GEMMs + FAVOR+ feature map ---
    // one RF^T B-fragment feeds 8 WMMAs (4 row tiles x {q,k})
#pragma unroll
    for (int cig = 0; cig < 2; ++cig) {
      const int ci = wave * 2 + cig;               // m column tile (0..15)
      v8f aq[4] = {}, ak[4] = {};
#pragma unroll
      for (int ks = 0; ks < 2; ++ks) {
        const v16bf bfr = frag(sRFT, Dh, ci * 16, ks * 32, lane);
#pragma unroll
        for (int ri = 0; ri < 4; ++ri) {
          aq[ri] = WMMA_BF16(frag(sQb, Dh, ri * 16, ks * 32, lane), bfr, aq[ri]);
          ak[ri] = WMMA_BF16(frag(sKb, Dh, ri * 16, ks * 32, lane), bfr, ak[ri]);
        }
      }
      const int col = ci * 16 + laneA;             // m index
#pragma unroll
      for (int ri = 0; ri < 4; ++ri) {
        const int r0 = ri * 16 + laneR;            // time base
        const v8f hkssq = *(const v8f*)(sHK + r0);
        v8bf pkT;
#pragma unroll
        for (int r = 0; r < 8; ++r) {
          const int row = r0 + r;
          const float qk = NC_F * (__expf(aq[ri][r]) + KEPS);
          const float kk = NC_F * (__expf(ak[ri][r] + fmaf(-0.5f * SCALE2, hkssq[r], khoff)) + KEPS);
          sQP[row * Mf + col] = (bf16)qk;
          sKP[row * Mf + col] = (bf16)kk;
          pkT[r] = (bf16)kk;
        }
        *(v8bf*)(sKPT + col * Dh + r0) = pkT;
      }
    }
    __syncthreads();

    // --- den0 = q' . Z_prev ---
    if (tid < 64) {
      float s = 0.f;
      for (int jm = 0; jm < Mf; ++jm) s += (float)sQP[tid * Mf + jm] * sZ[jm];
      sDen[tid] = s;
    }

    // --- fused: A = q'k'^T and O_inter = q'S_prev (shared q' A-fragments) ---
    v8f accA0 = {}, accA1 = {}, accO0 = {}, accO1 = {};
#pragma unroll
    for (int ks = 0; ks < 8; ++ks) {
      const v16bf af = frag(sQP, Mf, riO * 16, ks * 32, lane);
      accA0 = WMMA_BF16(af, frag(sKP,  Mf, ci0 * 16, ks * 32, lane), accA0);
      accA1 = WMMA_BF16(af, frag(sKP,  Mf, ci1 * 16, ks * 32, lane), accA1);
      accO0 = WMMA_BF16(af, frag(sSbT, Mf, ci0 * 16, ks * 32, lane), accO0);
      accO1 = WMMA_BF16(af, frag(sSbT, Mf, ci1 * 16, ks * 32, lane), accO1);
    }
    {
      const int r0 = riO * 16 + laneR;
#pragma unroll
      for (int r = 0; r < 8; ++r) {
        sA[(r0 + r) * Cc + ci0 * 16 + laneA] = accA0[r];
        sA[(r0 + r) * Cc + ci1 * 16 + laneA] = accA1[r];
      }
    }
    __syncthreads();

    // --- causal mask, row-sum into den, bf16 convert ---
    {
      float part = 0.f;
      v8bf m0, m1;
#pragma unroll
      for (int i = 0; i < 8; ++i) {
        const int colm = q4 * 16 + i;
        const float v = (colm <= row1) ? sA[row1 * Cc + colm] : 0.f;
        m0[i] = (bf16)v;  part += v;
      }
#pragma unroll
      for (int i = 0; i < 8; ++i) {
        const int colm = q4 * 16 + 8 + i;
        const float v = (colm <= row1) ? sA[row1 * Cc + colm] : 0.f;
        m1[i] = (bf16)v;  part += v;
      }
      *(v8bf*)(sAb + row1 * Cc + q4 * 16)     = m0;
      *(v8bf*)(sAb + row1 * Cc + q4 * 16 + 8) = m1;
      atomicAdd(&sDen[row1], part);
    }
    __syncthreads();

    // --- O += tril(A) V  (shared masked-A fragments), normalize, store ---
#pragma unroll
    for (int ks = 0; ks < 2; ++ks) {
      const v16bf af = frag(sAb, Cc, riO * 16, ks * 32, lane);
      accO0 = WMMA_BF16(af, frag(sVbT, Dh, ci0 * 16, ks * 32, lane), accO0);
      accO1 = WMMA_BF16(af, frag(sVbT, Dh, ci1 * 16, ks * 32, lane), accO1);
    }
    {
      const int r0 = riO * 16 + laneR;
      const v8f dv = *(const v8f*)(sDen + r0);
#pragma unroll
      for (int r = 0; r < 8; ++r) {
        float den = dv[r];
        if (fabsf(den) <= NSTAB) den += 2.f * NSTAB;
        float* orow = Out + ((size_t)(b * Ln + l0 + r0 + r)) * DM + hh * Dh;
        orow[ci0 * 16 + laneA] = accO0[r] / den;
        orow[ci1 * 16 + laneA] = accO1[r] / den;
      }
    }

    // --- S += k'^T V  (one k'^T A-fragment feeds 4 WMMAs) ---
#pragma unroll
    for (int mig = 0; mig < 2; ++mig) {
      const int mi = wave * 2 + mig;
#pragma unroll
      for (int ks = 0; ks < 2; ++ks) {
        const v16bf af = frag(sKPT, Dh, mi * 16, ks * 32, lane);
#pragma unroll
        for (int di = 0; di < 4; ++di)
          Sacc[mig * 4 + di] =
              WMMA_BF16(af, frag(sVbT, Dh, di * 16, ks * 32, lane), Sacc[mig * 4 + di]);
      }
    }

    // --- Z += colsum(k')  (contiguous rows of k'^T) ---
    {
      float s = 0.f;
#pragma unroll
      for (int t2 = 0; t2 < Cc; ++t2) s += (float)sKPT[tid * Dh + t2];
      sZ[tid] += s;
    }
    __syncthreads();
  }
}

// ---------------------------------------------------------------------------
extern "C" void kernel_launch(void* const* d_in, const int* in_sizes, int n_in,
                              void* d_out, int out_size, void* d_ws, size_t ws_size,
                              hipStream_t stream) {
  const float* Q  = (const float*)d_in[0];
  const float* Kg = (const float*)d_in[1];
  const float* V  = (const float*)d_in[2];
  const float* RF = (const float*)d_in[3];
  float* Out      = (float*)d_out;
  unsigned* minbits = (unsigned*)d_ws;

  (void)in_sizes; (void)n_in; (void)out_size; (void)ws_size;

  hipFuncSetAttribute((const void*)performer_chunk,
                      hipFuncAttributeMaxDynamicSharedMemorySize, SMEM_BYTES);

  performer_init_ws<<<1, 1, 0, stream>>>(minbits);
  performer_kstab<<<512, 256, 0, stream>>>(Kg, minbits);
  performer_chunk<<<dim3(Bn * Hn), dim3(256), SMEM_BYTES, stream>>>(Q, Kg, V, RF, minbits, Out);
}
